// NirvanaDetection_5772436046599
// MI455X (gfx1250) — matrix operand order repaired and checked by reading, not compiled, not run on hardware
//
#include <hip/hip_runtime.h>
#include <hip/hip_bf16.h>
#include <hip/hip_fp16.h>

typedef __attribute__((ext_vector_type(16))) _Float16 v16h;
typedef __attribute__((ext_vector_type(8)))  _Float16 v8h;
typedef __attribute__((ext_vector_type(8)))  float    v8f;

#define BN_EPS 1e-5f

// ---------------------------------------------------------------------------
// Prep: fold BN into scale/shift, build WMMA B-fragments for both convs,
// zero the score accumulator (every launch -> stateless / graph-safe).
// ---------------------------------------------------------------------------
__global__ __launch_bounds__(256) void prep_kernel(
    const float* __restrict__ w1, const float* __restrict__ b1,
    const float* __restrict__ g1, const float* __restrict__ be1,
    const float* __restrict__ m1, const float* __restrict__ v1,
    const float* __restrict__ w2, const float* __restrict__ b2,
    const float* __restrict__ g2, const float* __restrict__ be2,
    const float* __restrict__ m2, const float* __restrict__ v2,
    _Float16* __restrict__ bfrag1, _Float16* __restrict__ bfrag2,
    float* __restrict__ scale1, float* __restrict__ shift1,
    float* __restrict__ scale2, float* __restrict__ shift2,
    float* __restrict__ scoresum)
{
    int tid = threadIdx.x;
    if (tid < 32) {
        float inv = g1[tid] * rsqrtf(v1[tid] + BN_EPS);
        scale1[tid] = inv;
        shift1[tid] = b1[tid] * inv + be1[tid] - m1[tid] * inv;
    }
    if (tid < 64) {
        float inv = g2[tid] * rsqrtf(v2[tid] + BN_EPS);
        scale2[tid] = inv;
        shift2[tid] = b2[tid] * inv + be2[tid] - m2[tid] * inv;
    }
    // conv1 B fragments: 2 n-tiles x 32 lanes x 16 halves.
    // B layout (16-bit, 32x16): lane<16 -> K{0..7,16..23} in halves 0..15 (N=lane);
    //                           lane>=16 -> K{8..15,24..31} (N=lane-16).
    for (int e = tid; e < 2 * 32 * 16; e += 256) {
        int t    = e >> 9;
        int lane = (e >> 4) & 31;
        int i    = e & 15;
        int K    = ((lane < 16) ? 0 : 8) + i + ((i < 8) ? 0 : 8);     // K = c*9 + ky*3 + kx
        int n    = t * 16 + (lane & 15);
        bfrag1[e] = (K < 27) ? (_Float16)w1[n * 27 + K] : (_Float16)0.0f;
    }
    // conv2 B fragments: 9 K-chunks (r = ky*3+kx) x 4 n-tiles x 32 lanes x 16 halves.
    for (int e = tid; e < 9 * 4 * 32 * 16; e += 256) {
        int r    = e >> 11;
        int t    = (e >> 9) & 3;
        int lane = (e >> 4) & 31;
        int i    = e & 15;
        int c1   = ((lane < 16) ? 0 : 8) + i + ((i < 8) ? 0 : 8);     // channel within K-chunk
        int n    = t * 16 + (lane & 15);
        bfrag2[e] = (_Float16)w2[(n * 32 + c1) * 9 + r];
    }
    for (int e = tid; e < 512; e += 256) scoresum[e] = 0.0f;
}

// ---------------------------------------------------------------------------
// Layer 1: conv(3->32,s1,p1)+BN+ReLU+maxpool2 via WMMA implicit GEMM.
// Block = 128 thr (4 waves). Each block: 8x8 conv-pixel patch -> 4x4 of h.
// GEMM: M=16 pixels/wave, N=32 (2 tiles), K=27 padded to 32 (one chunk).
// h stored NHWC f16: [512][64][64][32].
// ---------------------------------------------------------------------------
__global__ __launch_bounds__(128) void conv1_kernel(
    const float* __restrict__ crops,
    const _Float16* __restrict__ bfrag1,
    const float* __restrict__ scale1, const float* __restrict__ shift1,
    _Float16* __restrict__ h)
{
    __shared__ _Float16 intile[3][10][10];   // 600 B input window (f16)
    __shared__ float    pbuf[64][32];        // 8 KB pre-pool activations

    int blk = blockIdx.x;
    int b   = blk >> 8;                      // image
    int pat = blk & 255;                     // 16x16 patches of 8x8 pixels
    int py0 = (pat >> 4) * 8;
    int px0 = (pat & 15) * 8;
    int tid = threadIdx.x;

    // Stage 10x10x3 f32 window -> f16 LDS (zero-pad borders).
    for (int e = tid; e < 300; e += 128) {
        int c   = e / 100;
        int rem = e - c * 100;
        int ly  = rem / 10, lx = rem % 10;
        int gy  = py0 - 1 + ly, gx = px0 - 1 + lx;
        float val = 0.0f;
        if (gy >= 0 && gy < 128 && gx >= 0 && gx < 128)
            val = crops[((b * 3 + c) * 128 + gy) * 128 + gx];
        intile[c][ly][lx] = (_Float16)val;
    }
    __syncthreads();

    int wave = tid >> 5;
    int lane = tid & 31;
    int m_lo = lane & 15;                    // A-matrix row this lane feeds
    int py   = wave * 2 + (m_lo >> 3);       // pixel in 8x8 patch (m = py*8+px)
    int px   = m_lo & 7;
    int h0   = (lane < 16) ? 0 : 8;          // A layout half-select

    // Build A fragment: lane<16 holds K{0..7,16..23}, lane>=16 K{8..15,24..31}.
    v16h a;
#pragma unroll
    for (int i = 0; i < 16; ++i) {
        int K  = h0 + i + ((i < 8) ? 0 : 8);
        int Kc = (K < 27) ? K : 26;          // clamp, select 0 after load
        int c  = Kc / 9;
        int r  = Kc - c * 9;
        _Float16 v = intile[c][py + r / 3][px + (r % 3)];
        a[i] = (K < 27) ? v : (_Float16)0.0f;
    }

    v16h bf0 = *(const v16h*)(bfrag1 + (0 * 32 + lane) * 16);
    v16h bf1 = *(const v16h*)(bfrag1 + (1 * 32 + lane) * 16);
    v8f acc0 = {};
    v8f acc1 = {};
    acc0 = __builtin_amdgcn_wmma_f32_16x16x32_f16(false, a, false, bf0, (short)0, acc0, false, false);
    acc1 = __builtin_amdgcn_wmma_f32_16x16x32_f16(false, a, false, bf1, (short)0, acc1, false, false);

    // Epilogue: BN + ReLU into LDS (D layout: lane<16 -> M=v, lane>=16 -> M=v+8; N=lane&15).
    int n0 = lane & 15;
#pragma unroll
    for (int v = 0; v < 8; ++v) {
        int M = (lane < 16) ? v : v + 8;
        int m = wave * 16 + M;
        float x0 = acc0[v] * scale1[n0] + shift1[n0];
        float x1 = acc1[v] * scale1[n0 + 16] + shift1[n0 + 16];
        pbuf[m][n0]      = fmaxf(x0, 0.0f);
        pbuf[m][n0 + 16] = fmaxf(x1, 0.0f);
    }
    __syncthreads();

    // 2x2 maxpool -> h NHWC f16, packed pairs of channels (u32 stores).
    for (int e = tid; e < 256; e += 128) {
        int pos = e >> 4;                    // 0..15 pooled position
        int cp  = (e & 15) * 2;              // channel pair
        int qy = pos >> 2, qx = pos & 3;
        int m00 = qy * 16 + qx * 2;          // (2qy)*8 + 2qx
        float a0 = fmaxf(fmaxf(pbuf[m00][cp],     pbuf[m00 + 1][cp]),
                         fmaxf(pbuf[m00 + 8][cp], pbuf[m00 + 9][cp]));
        float a1 = fmaxf(fmaxf(pbuf[m00][cp + 1],     pbuf[m00 + 1][cp + 1]),
                         fmaxf(pbuf[m00 + 8][cp + 1], pbuf[m00 + 9][cp + 1]));
        int gy = (py0 >> 1) + qy;
        int gx = (px0 >> 1) + qx;
        union { _Float16 hh[2]; unsigned u; } pk;
        pk.hh[0] = (_Float16)a0;
        pk.hh[1] = (_Float16)a1;
        *(unsigned*)(h + (((b * 64 + gy) * 64 + gx) * 32 + cp)) = pk.u;
    }
}

// ---------------------------------------------------------------------------
// Layer 2: conv(32->64,s2,p1)+BN+sigmoid+maxpool2 + partial score sums.
// Block = 128 thr. Each block: one image, 8x8 conv2-output patch (of 32x32).
// GEMM: M=16 pixels/wave, N=64 (4 tiles), K=288 = 9 chunks of 32 channels.
// Tile staging uses gfx1250 async global->LDS b128 copies (ASYNCcnt);
// A fragments then come from the NHWC f16 LDS tile: 2 x ds_load_b128/chunk.
// ---------------------------------------------------------------------------
__global__ __launch_bounds__(128) void conv2_kernel(
    const _Float16* __restrict__ h,
    const _Float16* __restrict__ bfrag2,
    const float* __restrict__ scale2, const float* __restrict__ shift2,
    float* __restrict__ feat, float* __restrict__ scoresum)
{
    __shared__ _Float16 htile[18][18][32];   // 20736 B
    __shared__ float    pbuf[64][64];        // 16 KB pre-pool (raw acc)
    __shared__ float    red[4];

    int blk = blockIdx.x;
    int b   = blk >> 4;
    int pat = blk & 15;
    int oy0 = (pat >> 2) * 8;
    int ox0 = (pat & 3) * 8;
    int tid = threadIdx.x;

    // Warm GL2/WGP$ with the pre-swizzled B fragments (36 KB) while staging.
    __builtin_prefetch(bfrag2 + tid * 144, 0, 3);

    // Stage 17x17x32-halves input window (18x18 alloc) in 16B chunks.
    // In-bounds chunks: async global->LDS copy (no VGPR round-trip).
    // Out-of-bounds border chunks: plain LDS zero store (DScnt / barrier).
    for (int e = tid; e < 18 * 18 * 4; e += 128) {
        int posi = e >> 2;
        int ch   = (e & 3) * 8;
        int ly = posi / 18, lx = posi % 18;
        int y2 = 2 * oy0 - 1 + ly;
        int x2 = 2 * ox0 - 1 + lx;
        _Float16* dst = &htile[ly][lx][ch];
        if (y2 >= 0 && y2 < 64 && x2 >= 0 && x2 < 64) {
            const _Float16* src = h + ((b * 64 + y2) * 64 + x2) * 32 + ch;
            unsigned lds_addr = (unsigned)(size_t)dst;   // low 32b = LDS offset
            asm volatile("global_load_async_to_lds_b128 %0, %1, off"
                         :: "v"(lds_addr), "v"(src)
                         : "memory");
        } else {
            v8h z = {};
            *(v8h*)dst = z;
        }
    }
    asm volatile("s_wait_asynccnt 0x0" ::: "memory");
    __syncthreads();

    int wave = tid >> 5;
    int lane = tid & 31;
    int m_lo = lane & 15;
    int py   = wave * 2 + (m_lo >> 3);
    int px   = m_lo & 7;
    int h0   = (lane < 16) ? 0 : 8;

    v8f acc0 = {}, acc1 = {}, acc2 = {}, acc3 = {};
#pragma unroll
    for (int r = 0; r < 9; ++r) {
        int ky = r / 3, kx = r % 3;
        int ly = 2 * py + ky;                // in [0,17)
        int lx = 2 * px + kx;
        // A fragment: channels {h0..h0+7} and {16+h0..16+h0+7} -> 2x b128 LDS.
        v8h lo = *(const v8h*)&htile[ly][lx][h0];
        v8h hi = *(const v8h*)&htile[ly][lx][16 + h0];
        v16h a;
#pragma unroll
        for (int i = 0; i < 8; ++i) { a[i] = lo[i]; a[i + 8] = hi[i]; }

        const _Float16* bp = bfrag2 + ((r * 4 + 0) * 32 + lane) * 16;
        v16h bf0 = *(const v16h*)(bp);
        v16h bf1 = *(const v16h*)(bp + 512);
        v16h bf2 = *(const v16h*)(bp + 1024);
        v16h bf3 = *(const v16h*)(bp + 1536);
        acc0 = __builtin_amdgcn_wmma_f32_16x16x32_f16(false, a, false, bf0, (short)0, acc0, false, false);
        acc1 = __builtin_amdgcn_wmma_f32_16x16x32_f16(false, a, false, bf1, (short)0, acc1, false, false);
        acc2 = __builtin_amdgcn_wmma_f32_16x16x32_f16(false, a, false, bf2, (short)0, acc2, false, false);
        acc3 = __builtin_amdgcn_wmma_f32_16x16x32_f16(false, a, false, bf3, (short)0, acc3, false, false);
    }

    // Dump raw accumulators to LDS for cross-wave pooling.
    int n0 = lane & 15;
#pragma unroll
    for (int v = 0; v < 8; ++v) {
        int M = (lane < 16) ? v : v + 8;
        int m = wave * 16 + M;
        pbuf[m][n0]      = acc0[v];
        pbuf[m][n0 + 16] = acc1[v];
        pbuf[m][n0 + 32] = acc2[v];
        pbuf[m][n0 + 48] = acc3[v];
    }
    __syncthreads();

    // BN -> max-pool -> sigmoid (monotone: sigmoid(max)=max(sigmoid)) -> store.
    float lsum = 0.0f;
    for (int e = tid; e < 1024; e += 128) {
        int pos = e >> 6;                    // 0..15 pooled position
        int c   = e & 63;
        int qy = pos >> 2, qx = pos & 3;
        int m00 = qy * 16 + qx * 2;
        float sc = scale2[c], sh = shift2[c];
        float x00 = pbuf[m00][c]     * sc + sh;
        float x01 = pbuf[m00 + 1][c] * sc + sh;
        float x10 = pbuf[m00 + 8][c] * sc + sh;
        float x11 = pbuf[m00 + 9][c] * sc + sh;
        float mx  = fmaxf(fmaxf(x00, x01), fmaxf(x10, x11));
        float sg  = 1.0f / (1.0f + __expf(-mx));
        int gy = (oy0 >> 1) + qy;
        int gx = (ox0 >> 1) + qx;
        feat[((b * 64 + c) * 16 + gy) * 16 + gx] = sg;
        lsum += sg;
    }
    // Block reduction -> one atomic per block into scoresum[b].
#pragma unroll
    for (int off = 16; off > 0; off >>= 1)
        lsum += __shfl_down(lsum, off, 32);
    if (lane == 0) red[wave] = lsum;
    __syncthreads();
    if (tid == 0)
        atomicAdd(&scoresum[b], red[0] + red[1] + red[2] + red[3]);
}

// ---------------------------------------------------------------------------
// Finalize scores + detection threshold.
// ---------------------------------------------------------------------------
__global__ __launch_bounds__(256) void finalize_kernel(
    const float* __restrict__ scoresum,
    float* __restrict__ scores, float* __restrict__ detected)
{
    int i = blockIdx.x * blockDim.x + threadIdx.x;
    if (i < 512) {
        float s = scoresum[i] * (1.0f / 16384.0f);
        scores[i]   = s;
        detected[i] = (s >= 0.55f) ? 1.0f : 0.0f;
    }
}

extern "C" void kernel_launch(void* const* d_in, const int* in_sizes, int n_in,
                              void* d_out, int out_size, void* d_ws, size_t ws_size,
                              hipStream_t stream)
{
    const float* crops = (const float*)d_in[0];
    const float* w1  = (const float*)d_in[1];
    const float* b1  = (const float*)d_in[2];
    const float* g1  = (const float*)d_in[3];
    const float* be1 = (const float*)d_in[4];
    const float* m1  = (const float*)d_in[5];
    const float* v1  = (const float*)d_in[6];
    const float* w2  = (const float*)d_in[7];
    const float* b2  = (const float*)d_in[8];
    const float* g2  = (const float*)d_in[9];
    const float* be2 = (const float*)d_in[10];
    const float* m2  = (const float*)d_in[11];
    const float* v2  = (const float*)d_in[12];

    // Workspace layout (bytes):
    //   h (NHWC f16, 512x64x64x32) : 67,108,864
    //   bfrag1                     : 2,048
    //   bfrag2                     : 36,864
    //   scale1/shift1/scale2/shift2: 768
    //   scoresum (512 f32)         : 2,048
    char* ws = (char*)d_ws;
    _Float16* h       = (_Float16*)ws;
    _Float16* bfrag1  = (_Float16*)(ws + 67108864);
    _Float16* bfrag2  = (_Float16*)(ws + 67110912);
    float*    scale1  = (float*)(ws + 67147776);
    float*    shift1  = scale1 + 32;
    float*    scale2  = shift1 + 32;
    float*    shift2  = scale2 + 64;
    float*    scoresum = shift2 + 64;

    float* out      = (float*)d_out;
    float* feat     = out;                       // 512*64*16*16 = 8,388,608
    float* scores   = out + 8388608;             // 512
    float* detected = scores + 512;              // 512

    prep_kernel<<<dim3(1), dim3(256), 0, stream>>>(
        w1, b1, g1, be1, m1, v1, w2, b2, g2, be2, m2, v2,
        bfrag1, bfrag2, scale1, shift1, scale2, shift2, scoresum);

    conv1_kernel<<<dim3(512 * 256), dim3(128), 0, stream>>>(
        crops, bfrag1, scale1, shift1, h);

    conv2_kernel<<<dim3(512 * 16), dim3(128), 0, stream>>>(
        h, bfrag2, scale2, shift2, feat, scoresum);

    finalize_kernel<<<dim3(2), dim3(256), 0, stream>>>(scoresum, scores, detected);
}
